// Mamba_28991029248281
// MI455X (gfx1250) — matrix operand order repaired and checked by reading, not compiled, not run on hardware
//
#include <hip/hip_runtime.h>
#include <hip/hip_bf16.h>
#include <cstdint>

// ---------------- problem constants ----------------
constexpr int NLh = 8;            // layers
constexpr int Dh  = 768;          // hidden
constexpr int Eh  = 1536;         // intermediate
constexpr int SSN = 16;           // state dim N
constexpr int CK  = 4;            // conv kernel
constexpr int DTRh= 48;           // dt rank
constexpr int Bh  = 2;
constexpr int Lh  = 2048;
constexpr int Mtok = Bh * Lh;     // 4096 tokens
constexpr int NCH = 32;           // scan chunks
constexpr int CH  = Lh / NCH;     // 64 steps per chunk
constexpr float EPSF = 1e-5f;

// workspace layout (floats)
constexpr size_t BLD  = (size_t)Bh * Lh * Dh;        // 3,145,728
constexpr size_t BL2E = (size_t)Bh * Lh * 2 * Eh;    // 12,582,912
constexpr size_t BLE  = (size_t)Bh * Lh * Eh;        // 6,291,456
constexpr size_t BL80 = (size_t)Bh * Lh * 80;        // 327,680
constexpr size_t SLOC = (size_t)Bh * NCH * Eh * SSN; // 1,572,864
constexpr size_t SDTC = (size_t)Bh * NCH * Eh;       // 98,304

typedef __attribute__((ext_vector_type(16))) __bf16 v16bf;
typedef __attribute__((ext_vector_type(8)))  float  v8f;

__device__ __forceinline__ unsigned bf16pair(float lo, float hi) {
  unsigned a = __float_as_uint(lo);
  unsigned b = __float_as_uint(hi);
  a += 0x7fffu + ((a >> 16) & 1u);           // RNE
  b += 0x7fffu + ((b >> 16) & 1u);
  return (a >> 16) | (b & 0xffff0000u);
}

__device__ __forceinline__ float sigmoidf_(float x) {
  return 1.0f / (1.0f + __expf(-x));
}

// low 32 bits of a generic pointer to LDS == LDS byte offset (aperture in high bits)
__device__ __forceinline__ unsigned lds_byte_off(const void* p) {
  return (unsigned)(uintptr_t)p;
}

// async copy 16 bytes global -> LDS (GLOBAL_LOAD_ASYNC_TO_LDS_B128, tracked by ASYNCcnt)
__device__ __forceinline__ void async_cp16(unsigned lds_off, const void* gptr) {
  asm volatile("global_load_async_to_lds_b128 %0, %1, off"
               :: "v"(lds_off), "v"(gptr) : "memory");
}
__device__ __forceinline__ void async_wait0() {
  asm volatile("s_wait_asynccnt 0" ::: "memory");
}

// ---------------- RMSNorm ----------------
__global__ __launch_bounds__(256)
void rmsnorm_kernel(const float* __restrict__ x, const float* __restrict__ w,
                    float* __restrict__ out) {
  int row = blockIdx.x;
  const float* xr = x + (size_t)row * Dh;
  __shared__ float red[256];
  float s = 0.f;
  for (int d = threadIdx.x; d < Dh; d += 256) { float v = xr[d]; s += v * v; }
  red[threadIdx.x] = s;
  __syncthreads();
  for (int st = 128; st > 0; st >>= 1) {
    if (threadIdx.x < st) red[threadIdx.x] += red[threadIdx.x + st];
    __syncthreads();
  }
  float inv = rsqrtf(red[0] / (float)Dh + EPSF);
  for (int d = threadIdx.x; d < Dh; d += 256)
    out[(size_t)row * Dh + d] = xr[d] * inv * w[d];
}

// ---------------- WMMA bf16 GEMM: C[M,N] (+)= act(A[M,K] * Bw[N,K]^T + bias) ----------------
// flags: bit0 = accumulate into C (residual beta=1), bit1 = softplus epilogue
__global__ __launch_bounds__(256)
void gemm_wmma_bf16(const float* __restrict__ A, int lda,
                    const float* __restrict__ Bw,
                    const float* __restrict__ bias,
                    float* __restrict__ C, int ldc,
                    int M, int N, int K, int flags) {
  __shared__ unsigned Als[128][17];   // 128 rows x 32 K (bf16 pairs), +1 pad
  __shared__ unsigned Bls[128][17];

  const int blockN = blockIdx.x, blockM = blockIdx.y;
  const int tid  = threadIdx.x;
  const int wid  = tid >> 5, lane = tid & 31;
  const int wm   = wid >> 2, wn = wid & 3;     // waves: 2 x 4
  const int lm   = lane & 15, half = lane >> 4;

  const bool fullMN = (blockM * 128 + 128 <= M) && (blockN * 128 + 128 <= N);

  union Frag { unsigned u[8]; v16bf v; };
  v8f acc[4][2];
#pragma unroll
  for (int i = 0; i < 4; ++i)
#pragma unroll
    for (int j = 0; j < 2; ++j)
#pragma unroll
      for (int v = 0; v < 8; ++v) acc[i][j][v] = 0.f;

  for (int k0 = 0; k0 < K; k0 += 32) {
    if (fullMN && (k0 + 32 <= K)) {
      // fast path: unguarded vectorized staging (global_load_b64, fully coalesced)
#pragma unroll
      for (int it = 0; it < 8; ++it) {
        int p = tid + it * 256;            // 0..2047
        int row = p >> 4, cp = p & 15;
        const float2 av = *(const float2*)(A  + (size_t)(blockM * 128 + row) * lda + k0 + cp * 2);
        Als[row][cp] = bf16pair(av.x, av.y);
        const float2 bv = *(const float2*)(Bw + (size_t)(blockN * 128 + row) * K   + k0 + cp * 2);
        Bls[row][cp] = bf16pair(bv.x, bv.y);
      }
      if (k0 + 32 < K) {                   // prefetch next k-tile (global_prefetch_b8)
        int row = tid >> 1, half2 = tid & 1;
        __builtin_prefetch(A  + (size_t)(blockM * 128 + row) * lda + k0 + 32 + half2 * 16, 0, 1);
        __builtin_prefetch(Bw + (size_t)(blockN * 128 + row) * K   + k0 + 32 + half2 * 16, 0, 1);
      }
    } else {
      // guarded path for edge tiles (N=80 block, K=48 tail): zero-pad OOB
#pragma unroll
      for (int it = 0; it < 8; ++it) {
        int p = tid + it * 256;
        int row = p >> 4, cp = p & 15;
        int k = k0 + cp * 2;
        int gr = blockM * 128 + row;
        float lo = (gr < M && k     < K) ? A[(size_t)gr * lda + k]     : 0.f;
        float hi = (gr < M && k + 1 < K) ? A[(size_t)gr * lda + k + 1] : 0.f;
        Als[row][cp] = bf16pair(lo, hi);
        int gn = blockN * 128 + row;
        lo = (gn < N && k     < K) ? Bw[(size_t)gn * K + k]     : 0.f;
        hi = (gn < N && k + 1 < K) ? Bw[(size_t)gn * K + k + 1] : 0.f;
        Bls[row][cp] = bf16pair(lo, hi);
      }
    }
    __syncthreads();

    Frag af[4], bfr[2];
#pragma unroll
    for (int i = 0; i < 4; ++i) {
      int r = wm * 64 + i * 16 + lm;     // A-frag: lane holds row m, 16 K-values
#pragma unroll
      for (int v = 0; v < 8; ++v)
        af[i].u[v] = Als[r][(v < 4) ? (half * 4 + v) : (8 + half * 4 + (v - 4))];
    }
#pragma unroll
    for (int j = 0; j < 2; ++j) {
      int r = wn * 32 + j * 16 + lm;     // B-frag: lane holds col n, 16 K-values
#pragma unroll
      for (int v = 0; v < 8; ++v)
        bfr[j].u[v] = Bls[r][(v < 4) ? (half * 4 + v) : (8 + half * 4 + (v - 4))];
    }
#pragma unroll
    for (int i = 0; i < 4; ++i)
#pragma unroll
      for (int j = 0; j < 2; ++j)
        acc[i][j] = __builtin_amdgcn_wmma_f32_16x16x32_bf16(
            false, af[i].v, false, bfr[j].v, (short)0, acc[i][j], false, false);
    __syncthreads();
  }

  // epilogue
#pragma unroll
  for (int i = 0; i < 4; ++i) {
#pragma unroll
    for (int j = 0; j < 2; ++j) {
      int col = blockN * 128 + wn * 32 + j * 16 + lm;
      if (col < N) {
        float bv = bias ? bias[col] : 0.f;
#pragma unroll
        for (int v = 0; v < 8; ++v) {
          int row = blockM * 128 + wm * 64 + i * 16 + half * 8 + v;
          if (row < M) {
            float val = acc[i][j][v] + bv;
            if (flags & 2) val = (val > 20.f) ? val : log1pf(__expf(val));
            size_t ci = (size_t)row * ldc + col;
            if (flags & 1) val += C[ci];
            C[ci] = val;
          }
        }
      }
    }
  }
}

// ---------------- causal depthwise conv (K=4) + SiLU ----------------
__global__ __launch_bounds__(256)
void conv_silu_kernel(const float* __restrict__ proj, const float* __restrict__ cw,
                      const float* __restrict__ cb, float* __restrict__ u) {
  int idx = blockIdx.x * 256 + threadIdx.x;
  if (idx >= Bh * Lh * Eh) return;
  int e = idx % Eh;
  int t = (idx / Eh) % Lh;
  int b = idx / (Eh * Lh);
  const float* pr = proj + (size_t)b * Lh * (2 * Eh) + e;
  float acc = cb[e];
#pragma unroll
  for (int k = 0; k < CK; ++k) {
    int tau = t - (CK - 1) + k;
    if (tau >= 0) acc += cw[e * CK + k] * pr[(size_t)tau * (2 * Eh)];
  }
  u[idx] = acc * sigmoidf_(acc);
}

// ---------------- selective scan, chunked (3 phases) ----------------
// grid: Bh * NCH * (Eh/256), 256 threads; thread owns one (b, chunk, e)
__global__ __launch_bounds__(256)
void scan_phaseA(const float* __restrict__ ssm, const float* __restrict__ dt,
                 const float* __restrict__ u, const float* __restrict__ A_log,
                 float* __restrict__ s_loc, float* __restrict__ sdt_buf) {
  const int ebks = Eh / 256;
  int bid = blockIdx.x;
  int eb = bid % ebks, ch = (bid / ebks) % NCH, b = bid / (ebks * NCH);
  int e = eb * 256 + threadIdx.x;

  __shared__ float Bs[CH][SSN];
  {
    // async global->LDS: 256 threads x 16B = 4KB tile of B_t[n]
    int st = threadIdx.x >> 2, seg = threadIdx.x & 3;          // row, 16B segment
    unsigned lo = lds_byte_off(&Bs[0][0]) + (unsigned)((st * SSN + seg * 4) * 4);
    const float* gp = ssm + ((size_t)(b * Lh + ch * CH + st)) * 80 + 48 + seg * 4;
    async_cp16(lo, gp);
    async_wait0();
  }
  __syncthreads();

  float a[SSN], state[SSN];
#pragma unroll
  for (int n = 0; n < SSN; ++n) { a[n] = -__expf(A_log[e * SSN + n]); state[n] = 0.f; }
  float sdt = 0.f;
  for (int tl = 0; tl < CH; ++tl) {
    size_t base = ((size_t)(b * Lh + ch * CH + tl)) * Eh + e;
    float dtv = dt[base], uv = u[base];
    float dtu = dtv * uv;
    sdt += dtv;
#pragma unroll
    for (int n = 0; n < SSN; ++n)
      state[n] = state[n] * __expf(dtv * a[n]) + dtu * Bs[tl][n];
  }
  size_t sIdx = ((size_t)(b * NCH + ch)) * Eh + e;
#pragma unroll
  for (int n = 0; n < SSN; ++n) s_loc[sIdx * SSN + n] = state[n];
  sdt_buf[sIdx] = sdt;
}

// grid: (Bh*Eh)/256; thread owns one (b,e); 32-step sequential combine
__global__ __launch_bounds__(256)
void scan_phaseB(const float* __restrict__ A_log, const float* __restrict__ s_loc,
                 const float* __restrict__ sdt_buf, float* __restrict__ s_init) {
  int idx = blockIdx.x * 256 + threadIdx.x;
  int e = idx % Eh, b = idx / Eh;
  float a[SSN], s[SSN];
#pragma unroll
  for (int n = 0; n < SSN; ++n) { a[n] = -__expf(A_log[e * SSN + n]); s[n] = 0.f; }
  for (int c = 0; c < NCH; ++c) {
    size_t sIdx = ((size_t)(b * NCH + c)) * Eh + e;
#pragma unroll
    for (int n = 0; n < SSN; ++n) s_init[sIdx * SSN + n] = s[n];
    float Sd = sdt_buf[sIdx];
#pragma unroll
    for (int n = 0; n < SSN; ++n)
      s[n] = s[n] * __expf(a[n] * Sd) + s_loc[sIdx * SSN + n];
  }
}

// rerun each chunk seeded with its true initial state; fuse D-skip + gate(SiLU).
// writes gated y in place over u.
__global__ __launch_bounds__(256)
void scan_phaseC(const float* __restrict__ ssm, const float* __restrict__ dt,
                 float* __restrict__ u, const float* __restrict__ proj,
                 const float* __restrict__ A_log, const float* __restrict__ Dp,
                 const float* __restrict__ s_init) {
  const int ebks = Eh / 256;
  int bid = blockIdx.x;
  int eb = bid % ebks, ch = (bid / ebks) % NCH, b = bid / (ebks * NCH);
  int e = eb * 256 + threadIdx.x;

  __shared__ float Bs[CH][SSN], Cs[CH][SSN];
  {
    // async global->LDS: B tile and C tile (2 x 4KB)
    int st = threadIdx.x >> 2, seg = threadIdx.x & 3;
    size_t rb = ((size_t)(b * Lh + ch * CH + st)) * 80;
    unsigned lb = lds_byte_off(&Bs[0][0]) + (unsigned)((st * SSN + seg * 4) * 4);
    unsigned lc = lds_byte_off(&Cs[0][0]) + (unsigned)((st * SSN + seg * 4) * 4);
    async_cp16(lb, ssm + rb + 48 + seg * 4);
    async_cp16(lc, ssm + rb + 64 + seg * 4);
    async_wait0();
  }
  __syncthreads();

  size_t sIdx = ((size_t)(b * NCH + ch)) * Eh + e;
  float a[SSN], state[SSN];
#pragma unroll
  for (int n = 0; n < SSN; ++n) {
    a[n] = -__expf(A_log[e * SSN + n]);
    state[n] = s_init[sIdx * SSN + n];
  }
  float dp = Dp[e];
  for (int tl = 0; tl < CH; ++tl) {
    size_t tok = (size_t)(b * Lh + ch * CH + tl);
    size_t base = tok * Eh + e;
    float dtv = dt[base], uv = u[base];
    float dtu = dtv * uv;
    float y = 0.f;
#pragma unroll
    for (int n = 0; n < SSN; ++n) {
      state[n] = state[n] * __expf(dtv * a[n]) + dtu * Bs[tl][n];
      y += state[n] * Cs[tl][n];
    }
    float g = proj[tok * (2 * Eh) + Eh + e];
    u[base] = (y + uv * dp) * (g * sigmoidf_(g));
  }
}

// ---------------- host orchestration ----------------
extern "C" void kernel_launch(void* const* d_in, const int* in_sizes, int n_in,
                              void* d_out, int out_size, void* d_ws, size_t ws_size,
                              hipStream_t stream) {
  const float* x        = (const float*)d_in[0];
  const float* norm_w   = (const float*)d_in[1];
  const float* in_w     = (const float*)d_in[2];
  const float* conv_w   = (const float*)d_in[3];
  const float* conv_b   = (const float*)d_in[4];
  const float* xproj_w  = (const float*)d_in[5];
  const float* dtproj_w = (const float*)d_in[6];
  const float* dtproj_b = (const float*)d_in[7];
  const float* A_log    = (const float*)d_in[8];
  const float* D_skip   = (const float*)d_in[9];
  const float* out_w    = (const float*)d_in[10];
  const float* fnorm_w  = (const float*)d_in[11];

  float* ws = (float*)d_ws;
  float* h      = ws;            // residual stream
  float* hn     = h      + BLD;
  float* proj   = hn     + BLD;
  float* u      = proj   + BL2E; // u, later gated y (in place)
  float* ssm    = u      + BLE;
  float* dtb    = ssm    + BL80;
  float* s_loc  = dtb    + BLE;
  float* sdt    = s_loc  + SLOC;
  float* s_init = sdt    + SDTC;

  hipMemcpyAsync(h, x, BLD * sizeof(float), hipMemcpyDeviceToDevice, stream);

  const dim3 g1((2 * Eh + 127) / 128, (Mtok + 127) / 128);   // in_proj  (24,32)
  const dim3 g2((80 + 127) / 128,     (Mtok + 127) / 128);   // x_proj   (1,32)
  const dim3 g3((Eh + 127) / 128,     (Mtok + 127) / 128);   // dt_proj  (12,32)
  const dim3 g4((Dh + 127) / 128,     (Mtok + 127) / 128);   // out_proj (6,32)
  const int scanBlocks = Bh * NCH * (Eh / 256);              // 384

  for (int l = 0; l < NLh; ++l) {
    rmsnorm_kernel<<<Mtok, 256, 0, stream>>>(h, norm_w + (size_t)l * Dh, hn);

    gemm_wmma_bf16<<<g1, 256, 0, stream>>>(hn, Dh, in_w + (size_t)l * 2 * Eh * Dh,
                                           nullptr, proj, 2 * Eh,
                                           Mtok, 2 * Eh, Dh, 0);

    conv_silu_kernel<<<(Bh * Lh * Eh + 255) / 256, 256, 0, stream>>>(
        proj, conv_w + (size_t)l * Eh * CK, conv_b + (size_t)l * Eh, u);

    gemm_wmma_bf16<<<g2, 256, 0, stream>>>(u, Eh, xproj_w + (size_t)l * 80 * Eh,
                                           nullptr, ssm, 80,
                                           Mtok, 80, Eh, 0);

    gemm_wmma_bf16<<<g3, 256, 0, stream>>>(ssm, 80, dtproj_w + (size_t)l * Eh * DTRh,
                                           dtproj_b + (size_t)l * Eh, dtb, Eh,
                                           Mtok, Eh, DTRh, 2 /*softplus*/);

    const float* Al = A_log + (size_t)l * Eh * SSN;
    scan_phaseA<<<scanBlocks, 256, 0, stream>>>(ssm, dtb, u, Al, s_loc, sdt);
    scan_phaseB<<<(Bh * Eh) / 256, 256, 0, stream>>>(Al, s_loc, sdt, s_init);
    scan_phaseC<<<scanBlocks, 256, 0, stream>>>(ssm, dtb, u, proj, Al,
                                                D_skip + (size_t)l * Eh, s_init);

    gemm_wmma_bf16<<<g4, 256, 0, stream>>>(u, Eh, out_w + (size_t)l * Dh * Eh,
                                           nullptr, h, Dh,
                                           Mtok, Dh, Eh, 1 /*residual*/);
  }

  rmsnorm_kernel<<<Mtok, 256, 0, stream>>>(h, fnorm_w, (float*)d_out);
}
